// QIKTMOE_43671227466081
// MI455X (gfx1250) — matrix operand order repaired and checked by reading, not compiled, not run on hardware
//
#include <hip/hip_runtime.h>
#include <math.h>

// ---------------------------------------------------------------------------
// Problem constants (from reference): B=32, T1=201 -> T=200, D=64, NQ=2048,
// NS=512, NE=16. Rows flattened as row = b*200 + t (6400 rows).
// ---------------------------------------------------------------------------
#define TT    200
#define ROWS  6400      // 32*200
#define GDIM  256       // 4*D
#define DD    64
#define NEXP  16

// bf16 weight-pool element offsets (all even; pairs never straddle segments)
#define OFF_EMB   0         // 4096*64
#define OFF_WIHQ  262144    // 256*64
#define OFF_WIHC  278528
#define OFF_WHHQ  294912
#define OFF_WHHC  311296
#define OFF_GWQ   327680    // 16*64
#define OFF_GWC   328704
#define OFF_MWQ   329728    // 16*64*64 (stored transposed [e][n][k])
#define OFF_MWC   395264
#define POOL_END  460800

typedef __attribute__((ext_vector_type(16))) __bf16 v16bf;
typedef __attribute__((ext_vector_type(8)))  float  v8f;

// Native f32 -> bf16 (hardware cvt, RNE)
__device__ inline __bf16 f2bf(float f) { return (__bf16)f; }

// A-fragment: 16x32 bf16 (MxK). lane L: m = L&15, K = ((L<16)?0:16) + j.
template <typename F>
__device__ inline v16bf load_a16(int lane, F&& src) {
  v16bf v;
  const int m  = lane & 15;
  const int kb = (lane < 16) ? 0 : 16;
#pragma unroll
  for (int j = 0; j < 16; ++j) {
    const int k = kb + j;
    v[j] = src(m, k);
  }
  return v;
}

// B-fragment: 32x16 bf16 (KxN). lane L: n = L&15, K = ((L<16)?0:16) + j.
template <typename F>
__device__ inline v16bf load_b16(int lane, F&& src) {
  v16bf v;
  const int n  = lane & 15;
  const int kb = (lane < 16) ? 0 : 16;
#pragma unroll
  for (int j = 0; j < 16; ++j) v[j] = src(kb + j, n);
  return v;
}

__device__ inline v8f wmma_bf16(v16bf a, v16bf b, v8f c) {
  return __builtin_amdgcn_wmma_f32_16x16x32_bf16(false, a, false, b,
                                                 (short)0, c, false, false);
}

__device__ inline float wave_sum(float v) {
#pragma unroll
  for (int off = 16; off > 0; off >>= 1) v += __shfl_xor(v, off, 32);
  return v;
}

__device__ inline float sigf(float x) { return 1.0f / (1.0f + __expf(-x)); }

// CDNA5 async copy: global -> LDS, 2x b128 per call (32 bytes), ASYNCcnt.
__device__ inline void async_ld_b256(unsigned lds_addr, unsigned long long gaddr) {
  asm volatile("global_load_async_to_lds_b128 %0, %1, off\n\t"
               "global_load_async_to_lds_b128 %0, %1, off offset:16"
               :: "v"(lds_addr), "v"(gaddr)
               : "memory");
}
__device__ inline void wait_async0() {
  asm volatile("s_wait_asynccnt 0" ::: "memory");
}

// ---------------------------------------------------------------------------
// Kernel 0: zero pred, emit r_shft = responses[:,1:] as float.
// ---------------------------------------------------------------------------
__global__ void init_out_kernel(const int* __restrict__ responses,
                                float* __restrict__ out) {
  int i = blockIdx.x * blockDim.x + threadIdx.x;
  if (i < ROWS) {
    out[i] = 0.0f;
    int b = i / TT, t = i % TT;
    out[ROWS + i] = (float)responses[b * (TT + 1) + t + 1];
  }
}

// ---------------------------------------------------------------------------
// Prep kernel: one-shot f32 -> bf16 conversion of all WMMA operand weights.
// mlp_w segments are transposed to [e][n][k] so B-fragment loads become
// contiguous 32-byte runs. 2 elements per thread. grid = (900), block = 256.
// ---------------------------------------------------------------------------
__global__ void prep_bf16_kernel(const float* __restrict__ emb,
                                 const float* __restrict__ wihq,
                                 const float* __restrict__ wihc,
                                 const float* __restrict__ whhq,
                                 const float* __restrict__ whhc,
                                 const float* __restrict__ gwq,
                                 const float* __restrict__ gwc,
                                 const float* __restrict__ mwq,
                                 const float* __restrict__ mwc,
                                 __bf16* __restrict__ pool) {
  const int e = (blockIdx.x * blockDim.x + threadIdx.x) * 2;
  if (e >= POOL_END) return;

  const float* src;
  int off;
  bool transpose = false;
  if      (e < OFF_WIHQ) { src = emb;  off = OFF_EMB; }
  else if (e < OFF_WIHC) { src = wihq; off = OFF_WIHQ; }
  else if (e < OFF_WHHQ) { src = wihc; off = OFF_WIHC; }
  else if (e < OFF_WHHC) { src = whhq; off = OFF_WHHQ; }
  else if (e < OFF_GWQ)  { src = whhc; off = OFF_WHHC; }
  else if (e < OFF_GWC)  { src = gwq;  off = OFF_GWQ; }
  else if (e < OFF_MWQ)  { src = gwc;  off = OFF_GWC; }
  else if (e < OFF_MWC)  { src = mwq;  off = OFF_MWQ; transpose = true; }
  else                   { src = mwc;  off = OFF_MWC; transpose = true; }

  const int j = e - off;
  if (!transpose) {
    pool[e]     = f2bf(src[j]);
    pool[e + 1] = f2bf(src[j + 1]);
  } else {
    // dst [e][n][k] <- src [e][k][n]; pairs contiguous in k.
#pragma unroll
    for (int u = 0; u < 2; ++u) {
      const int jj = j + u;
      const int ex = jj >> 12, rem = jj & 4095;
      const int n = rem >> 6, k = rem & 63;
      pool[e + u] = f2bf(src[(ex << 12) | (k << 6) | n]);
    }
  }
}

// ---------------------------------------------------------------------------
// Kernel 1: fused embedding gather + xg = xemb @ Wih^T + bih + bhh.
// All WMMA operands load bf16 directly. grid = (400, 16, 2), block = 32.
// ---------------------------------------------------------------------------
__global__ void xg_kernel(const __bf16* __restrict__ pool,
                          const float* __restrict__ bih_q,
                          const float* __restrict__ bhh_q,
                          const float* __restrict__ bih_c,
                          const float* __restrict__ bhh_c,
                          const int* __restrict__ skills,
                          const int* __restrict__ responses,
                          float* __restrict__ xg_q,
                          float* __restrict__ xg_c) {
  const int lstm = blockIdx.z;
  const __bf16* emb = pool + OFF_EMB;
  const __bf16* wih = pool + (lstm ? OFF_WIHC : OFF_WIHQ);
  const float* bih = lstm ? bih_c : bih_q;
  const float* bhh = lstm ? bhh_c : bhh_q;
  float* xg        = lstm ? xg_c  : xg_q;

  const int lane = threadIdx.x;
  const int row0 = blockIdx.x * 16;
  const int n0   = blockIdx.y * 16;

  // Embedding row for this lane's A-row (m = lane & 15)
  const int m   = lane & 15;
  const int row = row0 + m;
  const int b   = row / TT, t = row % TT;
  int s = skills[b * (TT + 1) + t];
  int r = responses[b * (TT + 1) + t];
  r = (r > -1) ? r : 0;                       // masked_r
  const __bf16* erow = emb + (size_t)(s + 2048 * r) * DD;

  v8f acc = {};
#pragma unroll
  for (int kc = 0; kc < 2; ++kc) {
    const int k0 = kc * 32;
    v16bf a  = load_a16(lane, [&](int /*mm*/, int k) { return erow[k0 + k]; });
    v16bf bf = load_b16(lane, [&](int k, int n) {
      return wih[(n0 + n) * DD + k0 + k];     // B = Wih^T (contiguous in k)
    });
    acc = wmma_bf16(a, bf, acc);
  }

  const int n    = lane & 15;
  const int gcol = n0 + n;
  const float bias = bih[gcol] + bhh[gcol];
  const int mb   = (lane < 16) ? 0 : 8;
#pragma unroll
  for (int rI = 0; rI < 8; ++rI) {
    const int rr = row0 + mb + rI;
    xg[(size_t)rr * GDIM + gcol] = acc[rI] + bias;
  }
}

// ---------------------------------------------------------------------------
// Kernel 2: sequential LSTM scan. One 1024-thread workgroup per LSTM.
// Whh (bf16, 32 KB) is pulled into LDS with async copies; xg tiles are
// double-buffered into LDS with async copies, prefetching step t+1 while
// step t computes. h lives in LDS as bf16 and streams out as bf16.
// grid = (1,1,2), block = 1024.
// ---------------------------------------------------------------------------
__global__ void lstm_kernel(const __bf16* __restrict__ pool,
                            const float* __restrict__ xg_q,
                            const float* __restrict__ xg_c,
                            __bf16* __restrict__ hbf_q,
                            __bf16* __restrict__ hbf_c) {
  const int lstm = blockIdx.z;
  const __bf16* whh = pool + (lstm ? OFF_WHHC : OFF_WHHQ);
  const float* xg   = lstm ? xg_c  : xg_q;
  __bf16* hbf       = lstm ? hbf_c : hbf_q;

  __shared__ __bf16 whh_s[GDIM * DD];     // Whh as bf16         (32 KB)
  __shared__ __bf16 hbf_s[32 * DD];       // hidden state (bf16) ( 4 KB)
  __shared__ float  c_s[32 * DD];         // cell state          ( 8 KB)
  __shared__ float  xg_s[2][32 * GDIM];   // double-buffered xg  (64 KB)

  const int tid = threadIdx.x;

  // Async-fill Whh into LDS: 32 bytes (16 bf16) per thread.
  async_ld_b256((unsigned)(size_t)&whh_s[tid * 16],
                (unsigned long long)(size_t)(whh + tid * 16));

  // Per-thread async-copy slice of xg: bi = tid>>5, 8 floats per thread.
  const int cp_bi = tid >> 5;
  const int cp_g0 = (tid & 31) * 8;
  const unsigned lds0 = (unsigned)(size_t)&xg_s[0][cp_bi * GDIM + cp_g0];
  const unsigned lds1 = (unsigned)(size_t)&xg_s[1][cp_bi * GDIM + cp_g0];
  const float* cp_base = xg + (size_t)cp_bi * TT * GDIM + cp_g0;

  // Prologue: async-load xg tile t=0 into buffer 0.
  async_ld_b256(lds0, (unsigned long long)(size_t)cp_base);

  for (int i = tid; i < 32 * DD; i += 1024) {
    hbf_s[i] = (__bf16)0.0f;
    c_s[i]   = 0.0f;
  }
  wait_async0();
  __syncthreads();

  const int lane  = tid & 31;
  const int wave  = tid >> 5;     // 0..31
  const int mtile = wave & 1;     // 2 tiles over batch (32 rows)
  const int ntile = wave >> 1;    // 16 tiles over 256 gate dims

  for (int t = 0; t < TT; ++t) {
    const int bn = t & 1;
    float* xgb = xg_s[bn];

    // Prefetch next step's xg tile into the other buffer (ASYNCcnt).
    if (t + 1 < TT) {
      async_ld_b256(bn ? lds0 : lds1,
                    (unsigned long long)(size_t)(cp_base + (size_t)(t + 1) * GDIM));
    }

    v8f acc = {};
#pragma unroll
    for (int kc = 0; kc < 2; ++kc) {
      const int k0 = kc * 32;
      v16bf a = load_a16(lane, [&](int mm, int k) {
        return hbf_s[(mtile * 16 + mm) * DD + k0 + k];
      });
      v16bf bf = load_b16(lane, [&](int k, int n) {
        return whh_s[(ntile * 16 + n) * DD + k0 + k];   // B = Whh^T
      });
      acc = wmma_bf16(a, bf, acc);
    }
    {
      const int n    = lane & 15;
      const int mb   = (lane < 16) ? 0 : 8;
      const int gcol = ntile * 16 + n;
#pragma unroll
      for (int rI = 0; rI < 8; ++rI) {
        const int bi = mtile * 16 + mb + rI;
        xgb[bi * GDIM + gcol] += acc[rI];   // g = xg + h@Whh^T, in place
      }
    }
    __syncthreads();

#pragma unroll
    for (int rep = 0; rep < 2; ++rep) {
      const int e  = tid + rep * 1024;   // 0..2047
      const int bi = e >> 6, d = e & 63;
      const float gi = xgb[bi * GDIM + d];
      const float gf = xgb[bi * GDIM + 64 + d];
      const float gg = xgb[bi * GDIM + 128 + d];
      const float go = xgb[bi * GDIM + 192 + d];
      const float c  = sigf(gf) * c_s[e] + sigf(gi) * tanhf(gg);
      const float h  = sigf(go) * tanhf(c);
      c_s[e] = c;
      const __bf16 hb = f2bf(h);
      hbf_s[e] = hb;
      hbf[((size_t)bi * TT + t) * DD + d] = hb;   // stream h out as bf16
    }
    wait_async0();        // next tile resident before it is consumed
    __syncthreads();
  }
}

// ---------------------------------------------------------------------------
// Kernel 3: MoE head. One block per 16-row tile per head; 16 waves = one
// wave per expert. A-fragments (h tile, bf16) loaded once and reused across
// the 4 N-tiles and the gate tile; B-fragments are contiguous bf16 runs
// (mlp_w pre-transposed). Output projection collapsed to a gathered 64-dot
// per expert. grid = (400, 2), block = 512.
// ---------------------------------------------------------------------------
__global__ void moe_kernel(const __bf16* __restrict__ pool,
                           const __bf16* __restrict__ hq,
                           const __bf16* __restrict__ hc,
                           const float* gb_q, const float* mb_q,
                           const float* ow_q, const float* ob_q,
                           const float* lg_q, const float* lb_q,
                           const float* gb_c, const float* mb_c,
                           const float* ow_c, const float* ob_c,
                           const float* lg_c, const float* lb_c,
                           const int* __restrict__ skills,
                           float* __restrict__ pred) {
  const int moe = blockIdx.y;
  const __bf16* h  = moe ? hc : hq;
  const __bf16* gw = pool + (moe ? OFF_GWC : OFF_GWQ);   // [n][k]
  const __bf16* mw = pool + (moe ? OFF_MWC : OFF_MWQ);   // [e][n][k] transposed
  const float* gb = moe ? gb_c : gb_q;
  const float* mb = moe ? mb_c : mb_q;
  const float* ow = moe ? ow_c : ow_q;
  const float* ob = moe ? ob_c : ob_q;
  const float* lg = moe ? lg_c : lg_q;
  const float* lb = moe ? lb_c : lb_q;
  const int out_dim = moe ? 512 : 2048;

  const int row0 = blockIdx.x * 16;
  const int lane = threadIdx.x & 31;
  const int wave = threadIdx.x >> 5;   // expert id 0..15

  __shared__ float z_s[NEXP * 16 * DD];   // [e][m][d]  (64 KB)
  __shared__ float gate_s[16 * NEXP];     // [m][e]
  __shared__ float contrib_s[16 * NEXP];  // [m][e]

  // h tile A-fragments: loaded once (bf16 straight from workspace).
  const v16bf a0 = load_a16(lane, [&](int mm, int k) {
    return h[(size_t)(row0 + mm) * DD + k];
  });
  const v16bf a1 = load_a16(lane, [&](int mm, int k) {
    return h[(size_t)(row0 + mm) * DD + 32 + k];
  });

  // --- per-expert MLP: z_e = h_tile(16x64) @ mlp_w[e](64x64) + mlp_b[e]
  {
    const __bf16* mw_e = mw + (size_t)wave * DD * DD;   // [n][k]
#pragma unroll
    for (int nt = 0; nt < 4; ++nt) {
      v8f acc = {};
      v16bf b0 = load_b16(lane, [&](int k, int n) {
        return mw_e[(nt * 16 + n) * DD + k];            // contiguous in k
      });
      acc = wmma_bf16(a0, b0, acc);
      v16bf b1 = load_b16(lane, [&](int k, int n) {
        return mw_e[(nt * 16 + n) * DD + 32 + k];
      });
      acc = wmma_bf16(a1, b1, acc);

      const int n   = lane & 15;
      const int mbv = (lane < 16) ? 0 : 8;
      const int col = nt * 16 + n;
      const float bias = mb[wave * DD + col];
#pragma unroll
      for (int rI = 0; rI < 8; ++rI)
        z_s[wave * (16 * DD) + (mbv + rI) * DD + col] = acc[rI] + bias;
    }
  }

  // --- gate logits: one WMMA tile by wave 0 (reuses a0/a1)
  if (wave == 0) {
    v8f acc = {};
    v16bf b0 = load_b16(lane, [&](int k, int n) { return gw[n * DD + k]; });
    acc = wmma_bf16(a0, b0, acc);
    v16bf b1 = load_b16(lane, [&](int k, int n) { return gw[n * DD + 32 + k]; });
    acc = wmma_bf16(a1, b1, acc);

    const int n   = lane & 15;
    const int mbv = (lane < 16) ? 0 : 8;
#pragma unroll
    for (int rI = 0; rI < 8; ++rI)
      gate_s[(mbv + rI) * NEXP + n] = acc[rI] + gb[n];
  }
  __syncthreads();

  // --- LayerNorm + ReLU on z_e (wave32 shuffle reductions, 2 elems/lane)
  {
    float* ze = &z_s[wave * (16 * DD)];
    const int d0 = 2 * lane;
#pragma unroll 4
    for (int m2 = 0; m2 < 16; ++m2) {
      const float v0 = ze[m2 * DD + d0];
      const float v1 = ze[m2 * DD + d0 + 1];
      const float s  = wave_sum(v0 + v1);
      const float ss = wave_sum(v0 * v0 + v1 * v1);
      const float mu  = s * (1.0f / 64.0f);
      const float var = ss * (1.0f / 64.0f) - mu * mu;
      const float rs  = rsqrtf(var + 1e-5f);
      float o0 = (v0 - mu) * rs * lg[d0] + lb[d0];
      float o1 = (v1 - mu) * rs * lg[d0 + 1] + lb[d0 + 1];
      ze[m2 * DD + d0]     = o0 > 0.0f ? o0 : 0.0f;
      ze[m2 * DD + d0 + 1] = o1 > 0.0f ? o1 : 0.0f;
    }
  }

  // --- softmax over experts (wave 0, one row per lane)
  if (wave == 0 && lane < 16) {
    float mx = -3.4e38f;
#pragma unroll
    for (int e = 0; e < NEXP; ++e) mx = fmaxf(mx, gate_s[lane * NEXP + e]);
    float sum = 0.0f;
#pragma unroll
    for (int e = 0; e < NEXP; ++e) {
      const float v = __expf(gate_s[lane * NEXP + e] - mx);
      gate_s[lane * NEXP + e] = v;
      sum += v;
    }
    const float inv = 1.0f / sum;
#pragma unroll
    for (int e = 0; e < NEXP; ++e) gate_s[lane * NEXP + e] *= inv;
  }
  __syncthreads();

  // --- gathered output projection: per row, one column of out_w[e]
  {
    const float* ze = &z_s[wave * (16 * DD)];
    const int d0 = 2 * lane;
    for (int m2 = 0; m2 < 16; ++m2) {
      const int row = row0 + m2;
      const int b = row / TT, t = row % TT;
      const int idx = skills[b * (TT + 1) + t + 1];   // q_shft
      const float p =
          ze[m2 * DD + d0]     * ow[((size_t)wave * DD + d0)     * out_dim + idx] +
          ze[m2 * DD + d0 + 1] * ow[((size_t)wave * DD + d0 + 1) * out_dim + idx];
      const float dot = wave_sum(p);
      if (lane == 0) {
        contrib_s[m2 * NEXP + wave] =
            gate_s[m2 * NEXP + wave] * (dot + ob[(size_t)wave * out_dim + idx]);
      }
    }
  }
  __syncthreads();

  // --- gated sum over experts, sigmoid, accumulate 0.5*y into pred
  if (wave == 0 && lane < 16) {
    float s = 0.0f;
#pragma unroll
    for (int e = 0; e < NEXP; ++e) s += contrib_s[lane * NEXP + e];
    const float y = sigf(s);
    atomicAdd(&pred[row0 + lane], 0.5f * y);
  }
}

// ---------------------------------------------------------------------------
extern "C" void kernel_launch(void* const* d_in, const int* in_sizes, int n_in,
                              void* d_out, int out_size, void* d_ws, size_t ws_size,
                              hipStream_t stream) {
  (void)in_sizes; (void)n_in; (void)out_size; (void)ws_size;

  const float* emb  = (const float*)d_in[0];
  const float* qWih = (const float*)d_in[1];
  const float* qWhh = (const float*)d_in[2];
  const float* qbih = (const float*)d_in[3];
  const float* qbhh = (const float*)d_in[4];
  const float* cWih = (const float*)d_in[5];
  const float* cWhh = (const float*)d_in[6];
  const float* cbih = (const float*)d_in[7];
  const float* cbhh = (const float*)d_in[8];
  const float* qgw  = (const float*)d_in[9];
  const float* qgb  = (const float*)d_in[10];
  const float* qmw  = (const float*)d_in[11];
  const float* qmb  = (const float*)d_in[12];
  const float* qow  = (const float*)d_in[13];
  const float* qob  = (const float*)d_in[14];
  const float* qlg  = (const float*)d_in[15];
  const float* qlb  = (const float*)d_in[16];
  const float* cgw  = (const float*)d_in[17];
  const float* cgb  = (const float*)d_in[18];
  const float* cmw  = (const float*)d_in[19];
  const float* cmb  = (const float*)d_in[20];
  const float* cow  = (const float*)d_in[21];
  const float* cob  = (const float*)d_in[22];
  const float* clg  = (const float*)d_in[23];
  const float* clb  = (const float*)d_in[24];
  const int* skills    = (const int*)d_in[25];
  const int* responses = (const int*)d_in[26];

  float* ws   = (float*)d_ws;
  float* xg_q = ws;                         // 6400*256 f32
  float* xg_c = xg_q + (size_t)ROWS * GDIM;
  __bf16* hbf_q = (__bf16*)(xg_c + (size_t)ROWS * GDIM);  // 6400*64 bf16
  __bf16* hbf_c = hbf_q + (size_t)ROWS * DD;
  __bf16* pool  = hbf_c + (size_t)ROWS * DD;              // bf16 weight pool

  float* out = (float*)d_out;

  init_out_kernel<<<dim3(25), dim3(256), 0, stream>>>(responses, out);

  prep_bf16_kernel<<<dim3((POOL_END / 2 + 255) / 256), dim3(256), 0, stream>>>(
      emb, qWih, cWih, qWhh, cWhh, qgw, cgw, qmw, cmw, pool);

  xg_kernel<<<dim3(400, 16, 2), dim3(32), 0, stream>>>(
      pool, qbih, qbhh, cbih, cbhh, skills, responses, xg_q, xg_c);

  lstm_kernel<<<dim3(1, 1, 2), dim3(1024), 0, stream>>>(
      pool, xg_q, xg_c, hbf_q, hbf_c);

  moe_kernel<<<dim3(400, 2), dim3(512), 0, stream>>>(
      pool, hbf_q, hbf_c,
      qgb, qmb, qow, qob, qlg, qlb,
      cgb, cmb, cow, cob, clg, clb,
      skills, out);
}
